// Attention_68616397521101
// MI455X (gfx1250) — compile-verified, matrix operand order checked by reading, not run on hardware
//
#include <hip/hip_runtime.h>
#include <hip/hip_bf16.h>

typedef __bf16 bf16;
typedef bf16  v16bf __attribute__((ext_vector_type(16)));
typedef float v8f   __attribute__((ext_vector_type(8)));

#define L_SEQ   2048
#define DMODEL  1024
#define NHEADS  16
#define DHEAD   64
#define P_LEN   2049
#define PE_PAD  2064

__device__ __forceinline__ v8f wmma_bf(v16bf a, v16bf b, v8f c) {
  // v_wmma_f32_16x16x32_bf16: (neg_a, A, neg_b, B, c_mod, C, reuse_a, reuse_b)
  return __builtin_amdgcn_wmma_f32_16x16x32_bf16(false, a, false, b, (short)0, c,
                                                 false, false);
}

// ---------------- elementwise prep ----------------
__global__ void k_f32_to_bf16(const float* __restrict__ x, bf16* __restrict__ y, int n) {
  int i = blockIdx.x * blockDim.x + threadIdx.x;
  if (i < n) y[i] = (bf16)x[i];
}

// r_kernel (H, D, dh) f32 -> rkT (H, dh, D) bf16
__global__ void k_transpose_rk(const float* __restrict__ rk, bf16* __restrict__ rkT) {
  int i = blockIdx.x * blockDim.x + threadIdx.x;
  if (i >= NHEADS * DMODEL * DHEAD) return;
  int h = i / (DMODEL * DHEAD);
  int r = i - h * (DMODEL * DHEAD);
  int m = r / DHEAD;
  int d = r - m * DHEAD;
  rkT[(size_t)h * DHEAD * DMODEL + (size_t)d * DMODEL + m] = (bf16)rk[i];
}

// qp (L, D) f32 -> qw/qr (H, L, dh) bf16 with per-head biases added
__global__ void k_post_q(const float* __restrict__ qp, const float* __restrict__ rwb,
                         const float* __restrict__ rrb, bf16* __restrict__ qw,
                         bf16* __restrict__ qr) {
  int i = blockIdx.x * blockDim.x + threadIdx.x;
  if (i >= L_SEQ * DMODEL) return;
  int l = i >> 10;
  int c = i & (DMODEL - 1);
  int h = c >> 6;
  int d = c & 63;
  float v = qp[i];
  size_t o = ((size_t)h * L_SEQ + l) * DHEAD + d;
  qw[o] = (bf16)(v + rwb[h * DHEAD + d]);
  qr[o] = (bf16)(v + rrb[h * DHEAD + d]);
}

// kp (L, D) f32 -> kh (H, L, dh) bf16
__global__ void k_post_k(const float* __restrict__ kp, bf16* __restrict__ kh) {
  int i = blockIdx.x * blockDim.x + threadIdx.x;
  if (i >= L_SEQ * DMODEL) return;
  int m = i >> 10;
  int c = i & (DMODEL - 1);
  int h = c >> 6;
  int d = c & 63;
  kh[((size_t)h * L_SEQ + m) * DHEAD + d] = (bf16)kp[i];
}

// vp (L, D) f32 -> vT (H, dh, L) bf16
__global__ void k_post_v(const float* __restrict__ vp, bf16* __restrict__ vT) {
  int i = blockIdx.x * blockDim.x + threadIdx.x;
  if (i >= L_SEQ * DMODEL) return;
  int m = i >> 10;
  int c = i & (DMODEL - 1);
  int h = c >> 6;
  int d = c & 63;
  vT[((size_t)h * DHEAD + d) * L_SEQ + m] = (bf16)vp[i];
}

// ---------------- generic NT GEMM: C(MxN) = A(MxK) * B(NxK)^T + bias ----------------
// one wave -> 32x64 output tile, double-buffered fragments; 4 waves / block
__global__ void k_gemm_nt(const bf16* __restrict__ A, const bf16* __restrict__ B,
                          const float* __restrict__ bias, float* __restrict__ C,
                          int M, int N, int K) {
  const int lane = threadIdx.x & 31;
  const int tile = blockIdx.x * 4 + (threadIdx.x >> 5);
  const int nt64 = N >> 6;
  if (tile >= (M >> 5) * nt64) return;
  const int m0 = (tile / nt64) << 5;
  const int n0 = (tile % nt64) << 6;
  const int row = lane & 15;
  const int kb  = (lane >> 4) << 4;
  const bf16* ap0 = A + (size_t)(m0 + row) * K + kb;
  const bf16* ap1 = A + (size_t)(m0 + 16 + row) * K + kb;
  const bf16* bp  = B + (size_t)(n0 + row) * K + kb;

  v8f c00 = {}, c01 = {}, c02 = {}, c03 = {};
  v8f c10 = {}, c11 = {}, c12 = {}, c13 = {};

  v16bf a0 = *(const v16bf*)(ap0);
  v16bf a1 = *(const v16bf*)(ap1);
  v16bf b0 = *(const v16bf*)(bp);
  v16bf b1 = *(const v16bf*)(bp + (size_t)16 * K);
  v16bf b2 = *(const v16bf*)(bp + (size_t)32 * K);
  v16bf b3 = *(const v16bf*)(bp + (size_t)48 * K);

  for (int k = 32; k < K; k += 32) {
    // prefetch next k-step while current WMMAs execute
    v16bf na0 = *(const v16bf*)(ap0 + k);
    v16bf na1 = *(const v16bf*)(ap1 + k);
    v16bf nb0 = *(const v16bf*)(bp + k);
    v16bf nb1 = *(const v16bf*)(bp + (size_t)16 * K + k);
    v16bf nb2 = *(const v16bf*)(bp + (size_t)32 * K + k);
    v16bf nb3 = *(const v16bf*)(bp + (size_t)48 * K + k);
    c00 = wmma_bf(a0, b0, c00);
    c01 = wmma_bf(a0, b1, c01);
    c02 = wmma_bf(a0, b2, c02);
    c03 = wmma_bf(a0, b3, c03);
    c10 = wmma_bf(a1, b0, c10);
    c11 = wmma_bf(a1, b1, c11);
    c12 = wmma_bf(a1, b2, c12);
    c13 = wmma_bf(a1, b3, c13);
    a0 = na0; a1 = na1;
    b0 = nb0; b1 = nb1; b2 = nb2; b3 = nb3;
  }
  c00 = wmma_bf(a0, b0, c00);
  c01 = wmma_bf(a0, b1, c01);
  c02 = wmma_bf(a0, b2, c02);
  c03 = wmma_bf(a0, b3, c03);
  c10 = wmma_bf(a1, b0, c10);
  c11 = wmma_bf(a1, b1, c11);
  c12 = wmma_bf(a1, b2, c12);
  c13 = wmma_bf(a1, b3, c13);

  const int hi = lane >> 4;
  for (int r = 0; r < 8; ++r) {
    const int rr0 = m0 + r + 8 * hi;
    const int rr1 = rr0 + 16;
    const int cc = n0 + (lane & 15);
    float* cp0 = C + (size_t)rr0 * N + cc;
    float* cp1 = C + (size_t)rr1 * N + cc;
    cp0[0]  = c00[r] + bias[cc];
    cp0[16] = c01[r] + bias[cc + 16];
    cp0[32] = c02[r] + bias[cc + 32];
    cp0[48] = c03[r] + bias[cc + 48];
    cp1[0]  = c10[r] + bias[cc];
    cp1[16] = c11[r] + bias[cc + 16];
    cp1[32] = c12[r] + bias[cc + 32];
    cp1[48] = c13[r] + bias[cc + 48];
  }
}

// ---------------- pe = pos_enc @ r_kernel  (per head, NT vs rkT) ----------------
// pe (H, PE_PAD, dh) bf16;  one wave -> 16 rows x 64 cols, double-buffered
__global__ void k_pe_gemm(const bf16* __restrict__ pos, const bf16* __restrict__ rkT,
                          bf16* __restrict__ pe) {
  const int lane = threadIdx.x & 31;
  const int tile = blockIdx.x * 4 + (threadIdx.x >> 5);
  const int PT = 129;  // ceil(2049/16)
  if (tile >= NHEADS * PT) return;
  const int h  = tile / PT;
  const int p0 = (tile % PT) << 4;
  const int row = lane & 15;
  const int kb  = (lane >> 4) << 4;
  int pr = p0 + row;
  if (pr > P_LEN - 1) pr = P_LEN - 1;  // clamp A rows for the partial last tile
  const bf16* ap = pos + (size_t)pr * DMODEL + kb;
  const bf16* bp = rkT + (size_t)h * DHEAD * DMODEL + (size_t)row * DMODEL + kb;

  v8f c0 = {}, c1 = {}, c2 = {}, c3 = {};
  v16bf a  = *(const v16bf*)(ap);
  v16bf b0 = *(const v16bf*)(bp);
  v16bf b1 = *(const v16bf*)(bp + (size_t)16 * DMODEL);
  v16bf b2 = *(const v16bf*)(bp + (size_t)32 * DMODEL);
  v16bf b3 = *(const v16bf*)(bp + (size_t)48 * DMODEL);
  for (int k = 32; k < DMODEL; k += 32) {
    v16bf na  = *(const v16bf*)(ap + k);
    v16bf nb0 = *(const v16bf*)(bp + k);
    v16bf nb1 = *(const v16bf*)(bp + (size_t)16 * DMODEL + k);
    v16bf nb2 = *(const v16bf*)(bp + (size_t)32 * DMODEL + k);
    v16bf nb3 = *(const v16bf*)(bp + (size_t)48 * DMODEL + k);
    c0 = wmma_bf(a, b0, c0);
    c1 = wmma_bf(a, b1, c1);
    c2 = wmma_bf(a, b2, c2);
    c3 = wmma_bf(a, b3, c3);
    a = na; b0 = nb0; b1 = nb1; b2 = nb2; b3 = nb3;
  }
  c0 = wmma_bf(a, b0, c0);
  c1 = wmma_bf(a, b1, c1);
  c2 = wmma_bf(a, b2, c2);
  c3 = wmma_bf(a, b3, c3);

  const int hi = lane >> 4;
  bf16* out = pe + (size_t)h * PE_PAD * DHEAD;
  for (int r = 0; r < 8; ++r) {
    int p = p0 + r + 8 * hi;           // < PE_PAD by construction
    int d = lane & 15;
    bf16* op = out + (size_t)p * DHEAD + d;
    op[0]  = (bf16)c0[r];
    op[16] = (bf16)c1[r];
    op[32] = (bf16)c2[r];
    op[48] = (bf16)c3[r];
  }
}

// ---------------- fused causal flash attention with rel-shift ----------------
// grid (L/64, H), block 128 (4 waves); each wave owns a 16-row query tile.
__global__ void k_attn(const bf16* __restrict__ qw, const bf16* __restrict__ qr,
                       const bf16* __restrict__ kh, const bf16* __restrict__ vT,
                       const bf16* __restrict__ pe, bf16* __restrict__ outbf) {
  __shared__ float srel[4][16 * 48];               // rel band per wave
  __shared__ __align__(64) bf16 spb[4][16 * 32];   // P tile staged to A-layout

  const int lane = threadIdx.x & 31;
  const int wave = threadIdx.x >> 5;
  const int h  = blockIdx.y;
  const int l0 = blockIdx.x * 64 + wave * 16;
  const int row = lane & 15;
  const int hi  = lane >> 4;
  const int kb  = hi << 4;
  const float scale = 0.125f;  // 1/sqrt(64)

  // preload query A-fragments (K=64 -> two K=32 chunks)
  const bf16* qwp = qw + ((size_t)h * L_SEQ + l0 + row) * DHEAD + kb;
  const bf16* qrp = qr + ((size_t)h * L_SEQ + l0 + row) * DHEAD + kb;
  v16bf aw0 = *(const v16bf*)(qwp);
  v16bf aw1 = *(const v16bf*)(qwp + 32);
  v16bf ar0 = *(const v16bf*)(qrp);
  v16bf ar1 = *(const v16bf*)(qrp + 32);

  const bf16* khh = kh + (size_t)h * L_SEQ * DHEAD;
  const bf16* vTh = vT + (size_t)h * DHEAD * L_SEQ;
  const bf16* peh = pe + (size_t)h * PE_PAD * DHEAD;

  v8f o0 = {}, o1 = {}, o2 = {}, o3 = {};
  float mrow[8], lrow[8];
  for (int r = 0; r < 8; ++r) { mrow[r] = -3.0e38f; lrow[r] = 0.0f; }

  float* rbuf = srel[wave];
  bf16*  pbuf = spb[wave];

  // preload K fragments for the first key block
  v16bf kc00, kc01, kc10, kc11;
  {
    const bf16* kp0 = khh + (size_t)(row) * DHEAD + kb;
    const bf16* kp1 = khh + (size_t)(16 + row) * DHEAD + kb;
    kc00 = *(const v16bf*)(kp0);
    kc01 = *(const v16bf*)(kp0 + 32);
    kc10 = *(const v16bf*)(kp1);
    kc11 = *(const v16bf*)(kp1 + 32);
  }

  const int nkb = (l0 + 16 + 31) >> 5;   // key blocks of 32 covering keys <= l0+15
  for (int kbi = 0; kbi < nkb; ++kbi) {
    const int m0 = kbi * 32;

    // ---- S = (q + r_w_bias) @ K^T : two 16x16 tiles (preloaded frags) ----
    v8f s0 = {}, s1 = {};
    s0 = wmma_bf(aw0, kc00, s0);
    s0 = wmma_bf(aw1, kc01, s0);
    s1 = wmma_bf(aw0, kc10, s1);
    s1 = wmma_bf(aw1, kc11, s1);

    // ---- prefetch next K tile (clamped; redundant on last iter) ----
    int mn = m0 + 32;
    if (mn > L_SEQ - 32) mn = L_SEQ - 32;
    const bf16* kn0 = khh + (size_t)(mn + row) * DHEAD + kb;
    const bf16* kn1 = khh + (size_t)(mn + 16 + row) * DHEAD + kb;
    v16bf nk00 = *(const v16bf*)(kn0);
    v16bf nk01 = *(const v16bf*)(kn0 + 32);
    v16bf nk10 = *(const v16bf*)(kn1);
    v16bf nk11 = *(const v16bf*)(kn1 + 32);

    // ---- prefetch V fragments for current block (consumed after softmax) ----
    const bf16* vp0 = vTh + (size_t)(row)      * L_SEQ + m0 + kb;
    const bf16* vp1 = vTh + (size_t)(16 + row) * L_SEQ + m0 + kb;
    const bf16* vp2 = vTh + (size_t)(32 + row) * L_SEQ + m0 + kb;
    const bf16* vp3 = vTh + (size_t)(48 + row) * L_SEQ + m0 + kb;
    v16bf bv0 = *(const v16bf*)vp0;
    v16bf bv1 = *(const v16bf*)vp1;
    v16bf bv2 = *(const v16bf*)vp2;
    v16bf bv3 = *(const v16bf*)vp3;

    // ---- rel band: rel_ext[i][j] = (q + r_r_bias)_i . pe[pbase + j], j in [0,48) ----
    const int pbase = L_SEQ - l0 - 15 + m0;  // >= 1 always
    for (int j = 0; j < 3; ++j) {
      int p = pbase + 16 * j + row;
      if (p > L_SEQ) p = L_SEQ;  // clamp: only masked elements read beyond
      const bf16* pp = peh + (size_t)p * DHEAD + kb;
      v16bf bp0 = *(const v16bf*)(pp);
      v16bf bp1 = *(const v16bf*)(pp + 32);
      v8f rc = {};
      rc = wmma_bf(ar0, bp0, rc);
      rc = wmma_bf(ar1, bp1, rc);
      for (int r = 0; r < 8; ++r)
        rbuf[(r + 8 * hi) * 48 + 16 * j + row] = rc[r];
    }
    asm volatile("s_wait_dscnt 0x0" ::: "memory");

    // ---- combine + causal mask + online softmax over 32 keys ----
    for (int r = 0; r < 8; ++r) {
      const int i  = r + 8 * hi;
      const int lq = l0 + i;
      const int jm0 = row, jm1 = row + 16;
      float rel0 = rbuf[i * 48 + jm0 - i + 15];  // shifted-rel gather
      float rel1 = rbuf[i * 48 + jm1 - i + 15];
      float v0 = (s0[r] + rel0) * scale;
      float v1 = (s1[r] + rel1) * scale;
      v0 = (m0 + jm0 <= lq) ? v0 : -1.0e30f;
      v1 = (m0 + jm1 <= lq) ? v1 : -1.0e30f;
      float mx = fmaxf(v0, v1);
      for (int d = 1; d < 16; d <<= 1) mx = fmaxf(mx, __shfl_xor(mx, d));
      float newm  = fmaxf(mrow[r], mx);
      float alpha = __expf(mrow[r] - newm);
      float p0 = __expf(v0 - newm);
      float p1 = __expf(v1 - newm);
      float ls = p0 + p1;
      for (int d = 1; d < 16; d <<= 1) ls += __shfl_xor(ls, d);
      lrow[r] = lrow[r] * alpha + ls;
      mrow[r] = newm;
      o0[r] *= alpha; o1[r] *= alpha; o2[r] *= alpha; o3[r] *= alpha;
      pbuf[i * 32 + jm0] = (bf16)p0;
      pbuf[i * 32 + jm1] = (bf16)p1;
    }
    asm volatile("s_wait_dscnt 0x0" ::: "memory");

    // ---- O += P @ V  (P: 16x32 A-frag from LDS; B fragments prefetched) ----
    v16bf pfrag = *(const v16bf*)(pbuf + row * 32 + kb);
    o0 = wmma_bf(pfrag, bv0, o0);
    o1 = wmma_bf(pfrag, bv1, o1);
    o2 = wmma_bf(pfrag, bv2, o2);
    o3 = wmma_bf(pfrag, bv3, o3);

    kc00 = nk00; kc01 = nk01; kc10 = nk10; kc11 = nk11;
  }

  // ---- epilogue: normalize, write bf16 attn output (L, D) row-major ----
  for (int r = 0; r < 8; ++r) {
    const int i = r + 8 * hi;
    const float inv = 1.0f / lrow[r];
    size_t base = (size_t)(l0 + i) * DMODEL + h * DHEAD + (lane & 15);
    outbf[base]      = (bf16)(o0[r] * inv);
    outbf[base + 16] = (bf16)(o1[r] * inv);
    outbf[base + 32] = (bf16)(o2[r] * inv);
    outbf[base + 48] = (bf16)(o3[r] * inv);
  }
}

// ---------------- host launcher ----------------
extern "C" void kernel_launch(void* const* d_in, const int* in_sizes, int n_in,
                              void* d_out, int out_size, void* d_ws, size_t ws_size,
                              hipStream_t stream) {
  (void)in_sizes; (void)n_in; (void)out_size; (void)ws_size;
  const float* q   = (const float*)d_in[0];
  const float* k   = (const float*)d_in[1];
  const float* v   = (const float*)d_in[2];
  const float* pos = (const float*)d_in[3];
  const float* Wq  = (const float*)d_in[4];
  const float* bq  = (const float*)d_in[5];
  const float* Wk  = (const float*)d_in[6];
  const float* bk  = (const float*)d_in[7];
  const float* Wv  = (const float*)d_in[8];
  const float* bv  = (const float*)d_in[9];
  const float* Wo  = (const float*)d_in[10];
  const float* bo  = (const float*)d_in[11];
  const float* rwb = (const float*)d_in[12];
  const float* rrb = (const float*)d_in[13];
  const float* rk  = (const float*)d_in[14];

  char* w = (char*)d_ws;
  size_t off = 0;
  auto alloc = [&](size_t bytes) -> void* {
    void* p = w + off;
    off += (bytes + 255) & ~(size_t)255;
    return p;
  };
  const size_t LD = (size_t)L_SEQ * DMODEL;  // 2M elements

  bf16* qbf  = (bf16*)alloc(LD * 2);
  bf16* kbf  = (bf16*)alloc(LD * 2);
  bf16* vbf  = (bf16*)alloc(LD * 2);
  bf16* wqb  = (bf16*)alloc((size_t)DMODEL * DMODEL * 2);
  bf16* wkb  = (bf16*)alloc((size_t)DMODEL * DMODEL * 2);
  bf16* wvb  = (bf16*)alloc((size_t)DMODEL * DMODEL * 2);
  bf16* wob  = (bf16*)alloc((size_t)DMODEL * DMODEL * 2);
  bf16* posb = (bf16*)alloc((size_t)P_LEN * DMODEL * 2);
  bf16* rkTb = (bf16*)alloc((size_t)NHEADS * DHEAD * DMODEL * 2);
  float* qpf = (float*)alloc(LD * 4);
  float* kpf = (float*)alloc(LD * 4);
  float* vpf = (float*)alloc(LD * 4);
  bf16* qwb  = (bf16*)alloc(LD * 2);
  bf16* qrb  = (bf16*)alloc(LD * 2);
  bf16* khb  = (bf16*)alloc(LD * 2);
  bf16* vTb  = (bf16*)alloc(LD * 2);
  bf16* peb  = (bf16*)alloc((size_t)NHEADS * PE_PAD * DHEAD * 2);
  bf16* attb = (bf16*)alloc(LD * 2);

  const int T = 256;
  const int nLD = (int)LD;
  const int nW  = DMODEL * DMODEL;
  const int nP  = P_LEN * DMODEL;
  const int nRK = NHEADS * DMODEL * DHEAD;

  k_f32_to_bf16<<<(nLD + T - 1) / T, T, 0, stream>>>(q, qbf, nLD);
  k_f32_to_bf16<<<(nLD + T - 1) / T, T, 0, stream>>>(k, kbf, nLD);
  k_f32_to_bf16<<<(nLD + T - 1) / T, T, 0, stream>>>(v, vbf, nLD);
  k_f32_to_bf16<<<(nW + T - 1) / T, T, 0, stream>>>(Wq, wqb, nW);
  k_f32_to_bf16<<<(nW + T - 1) / T, T, 0, stream>>>(Wk, wkb, nW);
  k_f32_to_bf16<<<(nW + T - 1) / T, T, 0, stream>>>(Wv, wvb, nW);
  k_f32_to_bf16<<<(nW + T - 1) / T, T, 0, stream>>>(Wo, wob, nW);
  k_f32_to_bf16<<<(nP + T - 1) / T, T, 0, stream>>>(pos, posb, nP);
  k_transpose_rk<<<(nRK + T - 1) / T, T, 0, stream>>>(rk, rkTb);

  // projections: 2048x1024x1024 NT GEMMs -> (M/32)*(N/64)=1024 wave tiles, 4/block
  k_gemm_nt<<<256, 128, 0, stream>>>(qbf, wqb, bq, qpf, L_SEQ, DMODEL, DMODEL);
  k_gemm_nt<<<256, 128, 0, stream>>>(kbf, wkb, bk, kpf, L_SEQ, DMODEL, DMODEL);
  k_gemm_nt<<<256, 128, 0, stream>>>(vbf, wvb, bv, vpf, L_SEQ, DMODEL, DMODEL);

  k_post_q<<<(nLD + T - 1) / T, T, 0, stream>>>(qpf, rwb, rrb, qwb, qrb);
  k_post_k<<<(nLD + T - 1) / T, T, 0, stream>>>(kpf, khb);
  k_post_v<<<(nLD + T - 1) / T, T, 0, stream>>>(vpf, vTb);

  // pe: 16 heads x 129 row-tiles = 2064 wave tiles -> 516 blocks
  k_pe_gemm<<<516, 128, 0, stream>>>(posb, rkTb, peb);

  // fused attention: grid (L/64, H)
  k_attn<<<dim3(L_SEQ / 64, NHEADS), 128, 0, stream>>>(qwb, qrb, khb, vTb, peb, attb);

  // output projection -> fp32 d_out
  k_gemm_nt<<<256, 128, 0, stream>>>(attb, wob, bo, (float*)d_out, L_SEQ, DMODEL, DMODEL);
}
